// AtomAttentionEncoder_35150012351223
// MI455X (gfx1250) — matrix operand order repaired and checked by reading, not compiled, not run on hardware
//
#include <hip/hip_runtime.h>
#include <hip/hip_fp16.h>

typedef __attribute__((ext_vector_type(16))) _Float16 v16h;
typedef __attribute__((ext_vector_type(8)))  float    v8f;
typedef __attribute__((ext_vector_type(4)))  unsigned u32x4;
typedef __attribute__((ext_vector_type(8)))  int      i32x8;
typedef __attribute__((ext_vector_type(4)))  int      i32x4;

#if defined(__has_builtin)
#  if __has_builtin(__builtin_amdgcn_tensor_load_to_lds)
#    define HAVE_TDM 1
#  endif
#endif
#ifndef HAVE_TDM
#  define HAVE_TDM 0
#endif

#define N_ATOMS 1536
#define N_TOK   192
#define C_ATOM  128
#define C_PAIR  16
#define C_TOKEN 384
#define C_ZPAIR 128
#define N_HEADS 4
#define DHEAD   32
#define NWIN    160   // padded band window: [i0-64, i0+96)

// ---------------------------------------------------------------------------
// WMMA GEMM, alignment-specialized: requires M%16==0 handled per-wave guard,
// N%16==0, K%32==0 (true for every call site here).
// 8 waves / block; one shared B panel per block; per-wave 16x16 C tile.
// ---------------------------------------------------------------------------
__global__ __launch_bounds__(256)
void k_gemm(const float* __restrict__ A, int lda,
            const float* __restrict__ B, int ldb,
            float* __restrict__ C, int ldc,
            int M, int N, int K,
            const float* __restrict__ bias,
            int act_a, int act_c, int beta) {
  __shared__ _Float16 As[8][16][32];
  __shared__ _Float16 Bs[16][32];   // Bs[n][k] = B[k][col0+n] (transposed)
  int wave = threadIdx.x >> 5, lane = threadIdx.x & 31;
  int col0 = blockIdx.x * 16;
  int row0 = (blockIdx.y * 8 + wave) * 16;
  bool active = row0 < M;
  int nl = lane & 15, hi = lane >> 4;
  v8f acc = {};
  int nk = K >> 5;
  for (int kc = 0; kc < nk; ++kc) {
    int kb0 = kc << 5;
    // cooperative B staging: 32k x 16n as float4 along n
    if (threadIdx.x < 128) {
      int k = threadIdx.x >> 2, nq = (threadIdx.x & 3) * 4;
      float4 bv = *(const float4*)(B + (size_t)(kb0 + k) * ldb + col0 + nq);
      Bs[nq + 0][k] = (_Float16)bv.x;
      Bs[nq + 1][k] = (_Float16)bv.y;
      Bs[nq + 2][k] = (_Float16)bv.z;
      Bs[nq + 3][k] = (_Float16)bv.w;
    }
    if (active) {
      // per-wave A staging: 16 rows x 32 cols, float4 along k
      for (int t = lane; t < 128; t += 32) {
        int m = t >> 3, kq = (t & 7) * 4;
        float4 av = *(const float4*)(A + (size_t)(row0 + m) * lda + kb0 + kq);
        if (act_a) {
          av.x = fmaxf(av.x, 0.f); av.y = fmaxf(av.y, 0.f);
          av.z = fmaxf(av.z, 0.f); av.w = fmaxf(av.w, 0.f);
        }
        As[wave][m][kq + 0] = (_Float16)av.x;
        As[wave][m][kq + 1] = (_Float16)av.y;
        As[wave][m][kq + 2] = (_Float16)av.z;
        As[wave][m][kq + 3] = (_Float16)av.w;
      }
    }
    __syncthreads();
    if (active) {
      v16h a, b;
#pragma unroll
      for (int v = 0; v < 8; ++v) {
        int ka = ((v >= 4) ? 16 : 0) + (hi ? 8 : 0) + (v & 3) * 2;
        a[2 * v]     = As[wave][nl][ka];
        a[2 * v + 1] = As[wave][nl][ka + 1];
        int kb = ((v >= 4) ? 16 : 0) + (hi ? 8 : 0) + (v & 3) * 2;
        b[2 * v]     = Bs[nl][kb];
        b[2 * v + 1] = Bs[nl][kb + 1];
      }
      acc = __builtin_amdgcn_wmma_f32_16x16x32_f16(false, a, false, b, (short)0,
                                                   acc, false, false);
    }
    __syncthreads();
  }
  if (active) {
#pragma unroll
    for (int v = 0; v < 8; ++v) {
      int m = row0 + hi * 8 + v;
      int n = col0 + nl;
      float val = acc[v];
      if (bias) val += bias[n];
      if (beta) val += C[(size_t)m * ldc + n];
      if (act_c == 1) val = val > 0.f ? val : 0.f;
      else if (act_c == 2) val = 1.f / (1.f + __expf(-val));
      C[(size_t)m * ldc + n] = val;
    }
  }
}

// ---------------------------------------------------------------------------
// Row LayerNorm: y = LN(x)*w + b  (w/b optional), one wave per row.
// ---------------------------------------------------------------------------
__global__ __launch_bounds__(256)
void k_ln_rows(const float* __restrict__ x, const float* __restrict__ w,
               const float* __restrict__ b, float* __restrict__ y,
               int rows, int width) {
  int row = blockIdx.x * 8 + (threadIdx.x >> 5);
  if (row >= rows) return;
  int lane = threadIdx.x & 31;
  const float* xr = x + (size_t)row * width;
  float s = 0.f, s2 = 0.f;
  for (int i = lane; i < width; i += 32) { float v = xr[i]; s += v; s2 += v * v; }
  for (int off = 16; off; off >>= 1) {
    s  += __shfl_xor(s,  off, 32);
    s2 += __shfl_xor(s2, off, 32);
  }
  float mu = s / width;
  float var = s2 / width - mu * mu;
  float inv = rsqrtf(var + 1e-5f);
  float* yr = y + (size_t)row * width;
  for (int i = lane; i < width; i += 32) {
    float v = (xr[i] - mu) * inv;
    if (w) v *= w[i];
    if (b) v += b[i];
    yr[i] = v;
  }
}

// ---------------------------------------------------------------------------
// Elementwise: 0: sigmoid(a)*b+c  1: silu(a)*b  2: sigmoid(a)*b  3: a+b  4: a*b
// ---------------------------------------------------------------------------
__global__ void k_eltwise(const float* __restrict__ a, const float* __restrict__ b,
                          const float* __restrict__ c, float* __restrict__ out,
                          int n, int mode) {
  int i = blockIdx.x * blockDim.x + threadIdx.x;
  if (i >= n) return;
  float r;
  if (mode == 0)      { float g = 1.f / (1.f + __expf(-a[i])); r = g * b[i] + c[i]; }
  else if (mode == 1) { float g = a[i] / (1.f + __expf(-a[i])); r = g * b[i]; }
  else if (mode == 2) { float g = 1.f / (1.f + __expf(-a[i])); r = g * b[i]; }
  else if (mode == 3) { r = a[i] + b[i]; }
  else                { r = a[i] * b[i]; }
  out[i] = r;
}

// ---------------------------------------------------------------------------
// Atom embedding: c = feats@w_embed + b_embed + ts[tok];  q0 = c + noisy@w_noisy
// ---------------------------------------------------------------------------
__global__ __launch_bounds__(128)
void k_embed(const float* __restrict__ pos, const float* __restrict__ charge,
             const float* __restrict__ amask, const float* __restrict__ elem,
             const float* __restrict__ names, const float* __restrict__ w_embed,
             const float* __restrict__ b_embed, const float* __restrict__ ts,
             const int* __restrict__ tok, const float* __restrict__ noisy,
             const float* __restrict__ w_noisy,
             float* __restrict__ c_out, float* __restrict__ q_out) {
  int a = blockIdx.x;
  int ch = threadIdx.x;
  const float* w = w_embed;  // [389][128]: pos(3),charge,mask,elem(128),names(256)
  float acc = b_embed[ch];
  acc += pos[a * 3 + 0] * w[0 * C_ATOM + ch];
  acc += pos[a * 3 + 1] * w[1 * C_ATOM + ch];
  acc += pos[a * 3 + 2] * w[2 * C_ATOM + ch];
  acc += charge[a] * w[3 * C_ATOM + ch];
  acc += amask[a] * w[4 * C_ATOM + ch];
  const float* we = w + 5 * C_ATOM;
  for (int i = 0; i < 128; ++i) acc += elem[a * 128 + i] * we[i * C_ATOM + ch];
  const float* wn = w + (5 + 128) * C_ATOM;
  for (int i = 0; i < 256; ++i) acc += names[a * 256 + i] * wn[i * C_ATOM + ch];
  acc += ts[(size_t)tok[a] * C_ATOM + ch];
  c_out[(size_t)a * C_ATOM + ch] = acc;
  float qv = acc;
  qv += noisy[a * 3 + 0] * w_noisy[0 * C_ATOM + ch];
  qv += noisy[a * 3 + 1] * w_noisy[1 * C_ATOM + ch];
  qv += noisy[a * 3 + 2] * w_noisy[2 * C_ATOM + ch];
  q_out[(size_t)a * C_ATOM + ch] = qv;
}

// ---------------------------------------------------------------------------
// Pair init: p[i,j,:] = (off@w_off + invd*w_d + w_m)*valid + zp[ti,tj] + row[i]+col[j]
// ---------------------------------------------------------------------------
__global__ __launch_bounds__(256)
void k_pair_init(const float* __restrict__ pos, const int* __restrict__ uid,
                 const int* __restrict__ tok, const float* __restrict__ w_off,
                 const float* __restrict__ w_d, const float* __restrict__ w_m,
                 const float* __restrict__ zp, const float* __restrict__ rowv,
                 const float* __restrict__ colv, float* __restrict__ p) {
  int i = blockIdx.x;
  int j = blockIdx.y * 256 + threadIdx.x;
  if (j >= N_ATOMS) return;
  float dx = pos[i * 3 + 0] - pos[j * 3 + 0];
  float dy = pos[i * 3 + 1] - pos[j * 3 + 1];
  float dz = pos[i * 3 + 2] - pos[j * 3 + 2];
  float invd = 1.f / (1.f + dx * dx + dy * dy + dz * dz);
  float valid = (uid[i] == uid[j]) ? 1.f : 0.f;
  const float* z = zp + ((size_t)tok[i] * N_TOK + tok[j]) * C_PAIR;
  float* pp = p + ((size_t)i * N_ATOMS + j) * C_PAIR;
#pragma unroll
  for (int ch = 0; ch < C_PAIR; ++ch) {
    float v = dx * w_off[0 * C_PAIR + ch] + dy * w_off[1 * C_PAIR + ch] +
              dz * w_off[2 * C_PAIR + ch] + invd * w_d[ch] + w_m[ch];
    v *= valid;
    v += z[ch] + rowv[(size_t)i * C_PAIR + ch] + colv[(size_t)j * C_PAIR + ch];
    pp[ch] = v;
  }
}

// ---------------------------------------------------------------------------
// Fused residual pair MLP: p += relu(relu(relu(p)@W1)@W2)@W3, 16 rows per wave.
// Tile of p staged global->LDS via Tensor Data Mover (TDM) when available.
// ---------------------------------------------------------------------------
__global__ __launch_bounds__(256)
void k_pair_mlp(float* __restrict__ p, const float* __restrict__ w1,
                const float* __restrict__ w2, const float* __restrict__ w3) {
  __shared__ _Float16 Ws[3][16][32];
  __shared__ _Float16 As[8][16][32];
  __shared__ float    Ptile[8][256];
  int wave = threadIdx.x >> 5, lane = threadIdx.x & 31;
  for (int t = threadIdx.x; t < 3 * 16 * 32; t += 256) {
    int i = t / 512, rem = t % 512, n = rem >> 5, k = rem & 31;
    const float* w = (i == 0) ? w1 : ((i == 1) ? w2 : w3);
    Ws[i][n][k] = (k < 16) ? (_Float16)w[k * 16 + n] : (_Float16)0.f;
  }
  size_t tile = (size_t)blockIdx.x * 8 + wave;
  float* pp = p + tile * 256;
#if HAVE_TDM
  {
    // D# group0: count=1, lds_addr, 57-bit global_addr, type=2 (image)
    unsigned lds_off = (unsigned)(size_t)&Ptile[wave][0];
    unsigned long long ga = (unsigned long long)(size_t)pp;
    u32x4 g0;
    g0[0] = 1u;
    g0[1] = lds_off;
    g0[2] = (unsigned)(ga & 0xffffffffu);
    g0[3] = (unsigned)((ga >> 32) & 0x1ffffffull) | (2u << 30);
    // D# group1: data_size=4B, tensor_dim0=256, tensor_dim1=1, tile_dim0=256,
    // tile_dim1=1, tensor_dim0_stride=256  (one contiguous 1KB row)
    i32x8 g1;
    g1[0] = 0x20000;              // data_size=2 (4 bytes), no mask/pad/iterate
    g1[1] = (int)(256u << 16);    // tensor_dim0[15:0] in bits 63:48
    g1[2] = (int)(1u << 16);      // tensor_dim0[31:16]=0, tensor_dim1[15:0]=1
    g1[3] = (int)(256u << 16);    // tensor_dim1[31:16]=0, tile_dim0=256
    g1[4] = 1;                    // tile_dim1=1, tile_dim2=0
    g1[5] = 256;                  // tensor_dim0_stride[31:0]
    g1[6] = 0;
    g1[7] = 0;
    i32x4 gz; gz[0] = 0; gz[1] = 0; gz[2] = 0; gz[3] = 0;
    i32x8 gz8; gz8[0] = 0; gz8[1] = 0; gz8[2] = 0; gz8[3] = 0;
    gz8[4] = 0; gz8[5] = 0; gz8[6] = 0; gz8[7] = 0;
    __builtin_amdgcn_tensor_load_to_lds(g0, g1, gz, gz, gz8, 0);
    __builtin_amdgcn_s_wait_tensorcnt(0);
  }
  for (int t = lane; t < 256; t += 32) {
    float v = Ptile[wave][t];
    int m = t >> 4, n = t & 15;
    As[wave][m][n] = (_Float16)fmaxf(v, 0.f);
    As[wave][m][16 + n] = (_Float16)0.f;
  }
#else
  __builtin_prefetch(pp + 256 * 8, 0, 1);
  for (int t = lane; t < 256; t += 32) {
    float v = pp[t];
    Ptile[wave][t] = v;
    int m = t >> 4, n = t & 15;
    As[wave][m][n] = (_Float16)fmaxf(v, 0.f);
    As[wave][m][16 + n] = (_Float16)0.f;
  }
#endif
  __syncthreads();
  int nl = lane & 15, hi = lane >> 4;
  v16h b0, b1, b2;
#pragma unroll
  for (int v = 0; v < 8; ++v) {
    int kb = ((v >= 4) ? 16 : 0) + (hi ? 8 : 0) + (v & 3) * 2;
    b0[2 * v] = Ws[0][nl][kb]; b0[2 * v + 1] = Ws[0][nl][kb + 1];
    b1[2 * v] = Ws[1][nl][kb]; b1[2 * v + 1] = Ws[1][nl][kb + 1];
    b2[2 * v] = Ws[2][nl][kb]; b2[2 * v + 1] = Ws[2][nl][kb + 1];
  }
  v8f acc = {};
#pragma unroll
  for (int stage = 0; stage < 3; ++stage) {
    v16h a;
#pragma unroll
    for (int v = 0; v < 8; ++v) {
      int ka = ((v >= 4) ? 16 : 0) + (hi ? 8 : 0) + (v & 3) * 2;
      a[2 * v]     = As[wave][nl][ka];
      a[2 * v + 1] = As[wave][nl][ka + 1];
    }
    v8f cz = {};
    acc = __builtin_amdgcn_wmma_f32_16x16x32_f16(
        false, a, false, (stage == 0) ? b0 : ((stage == 1) ? b1 : b2),
        (short)0, cz, false, false);
    __syncthreads();
    if (stage < 2) {
#pragma unroll
      for (int v = 0; v < 8; ++v) {
        float r = acc[v];
        As[wave][hi * 8 + v][nl] = (_Float16)(r > 0.f ? r : 0.f);
      }
      __syncthreads();
    }
  }
#pragma unroll
  for (int v = 0; v < 8; ++v) {
    int m = hi * 8 + v;
    pp[m * 16 + nl] = Ptile[wave][m * 16 + nl] + acc[v];   // residual
  }
}

// ---------------------------------------------------------------------------
// Banded attention (one wave per (query-tile, head)); pair bias LN fused.
// ---------------------------------------------------------------------------
__global__ __launch_bounds__(32)
void k_attn(const float* __restrict__ qh, const float* __restrict__ kh,
            const float* __restrict__ vh, const float* __restrict__ p,
            const float* __restrict__ lnp_w, const float* __restrict__ lnp_b,
            const float* __restrict__ w_pb, const float* __restrict__ amask,
            float* __restrict__ o) {
  __shared__ _Float16 Qs[16][32];
  __shared__ _Float16 Ks[NWIN][32];
  __shared__ _Float16 Vst[32][NWIN];   // Vst[d][key]
  __shared__ float    S[16][NWIN];
  __shared__ _Float16 Ps[16][NWIN];
  int lane = threadIdx.x;
  int qt = blockIdx.x, head = blockIdx.y;
  int i0 = qt * 16;
  int jbase = i0 - 64;
  for (int t = lane; t < 16 * 32; t += 32) {
    int m = t >> 5, k = t & 31;
    Qs[m][k] = (_Float16)qh[(size_t)(i0 + m) * C_ATOM + head * DHEAD + k];
  }
  for (int t = lane; t < NWIN * 32; t += 32) {
    int key = t >> 5, k = t & 31;
    int j = jbase + key;
    float kv = 0.f, vv = 0.f;
    if (j >= 0 && j < N_ATOMS) {
      kv = kh[(size_t)j * C_ATOM + head * DHEAD + k];
      vv = vh[(size_t)j * C_ATOM + head * DHEAD + k];
    }
    Ks[key][k] = (_Float16)kv;
    Vst[k][key] = (_Float16)vv;
  }
  // pair bias (LN over 16 channels) or -inf mask, cooperatively into S
  for (int t = lane; t < 16 * NWIN; t += 32) {
    int m = t / NWIN, key = t - m * NWIN;
    int i = i0 + m;
    int j = jbase + key;
    int d = i - j; if (d < 0) d = -d;
    float val = -1e30f;
    if (j >= 0 && j < N_ATOMS && d <= 64 && amask[j] > 0.5f) {
      const float4* pr = (const float4*)(p + ((size_t)i * N_ATOMS + j) * C_PAIR);
      float4 p0 = pr[0], p1 = pr[1], p2 = pr[2], p3 = pr[3];
      float tmp[16] = {p0.x, p0.y, p0.z, p0.w, p1.x, p1.y, p1.z, p1.w,
                       p2.x, p2.y, p2.z, p2.w, p3.x, p3.y, p3.z, p3.w};
      float s1 = 0.f, s2 = 0.f;
#pragma unroll
      for (int u = 0; u < 16; ++u) { s1 += tmp[u]; s2 += tmp[u] * tmp[u]; }
      float mu = s1 * (1.f / 16.f);
      float var = s2 * (1.f / 16.f) - mu * mu;
      float inv = rsqrtf(var + 1e-5f);
      float pb = 0.f;
#pragma unroll
      for (int u = 0; u < 16; ++u)
        pb += ((tmp[u] - mu) * inv * lnp_w[u] + lnp_b[u]) * w_pb[u * 4 + head];
      val = pb;
    }
    S[m][key] = val;
  }
  __syncthreads();
  int nl = lane & 15, hi = lane >> 4;
  const float scale = 0.17677669529663687f;  // 1/sqrt(32)
  for (int kt = 0; kt < NWIN / 16; ++kt) {
    v16h a, b;
#pragma unroll
    for (int v = 0; v < 8; ++v) {
      int ka = ((v >= 4) ? 16 : 0) + (hi ? 8 : 0) + (v & 3) * 2;
      a[2 * v] = Qs[nl][ka]; a[2 * v + 1] = Qs[nl][ka + 1];
      int kb = ((v >= 4) ? 16 : 0) + (hi ? 8 : 0) + (v & 3) * 2;
      b[2 * v] = Ks[kt * 16 + nl][kb]; b[2 * v + 1] = Ks[kt * 16 + nl][kb + 1];
    }
    v8f cz = {};
    v8f sacc = __builtin_amdgcn_wmma_f32_16x16x32_f16(false, a, false, b,
                                                      (short)0, cz, false, false);
#pragma unroll
    for (int v = 0; v < 8; ++v) {
      int m = hi * 8 + v;
      int col = kt * 16 + nl;
      float bias = S[m][col];
      if (bias > -1e29f) S[m][col] = sacc[v] * scale + bias;
    }
  }
  __syncthreads();
  if (lane < 16) {
    int r = lane;
    float mx = -1e30f;
    for (int j2 = 0; j2 < NWIN; ++j2) mx = fmaxf(mx, S[r][j2]);
    float sum = 0.f;
    for (int j2 = 0; j2 < NWIN; ++j2) { float e = __expf(S[r][j2] - mx); S[r][j2] = e; sum += e; }
    float inv = 1.f / sum;
    for (int j2 = 0; j2 < NWIN; ++j2) Ps[r][j2] = (_Float16)(S[r][j2] * inv);
  }
  __syncthreads();
  for (int ndh = 0; ndh < 2; ++ndh) {
    v8f acc = {};
    for (int kc = 0; kc < NWIN / 32; ++kc) {
      v16h a, b;
#pragma unroll
      for (int v = 0; v < 8; ++v) {
        int ka = ((v >= 4) ? 16 : 0) + (hi ? 8 : 0) + (v & 3) * 2;
        a[2 * v] = Ps[nl][kc * 32 + ka]; a[2 * v + 1] = Ps[nl][kc * 32 + ka + 1];
        int kb = ((v >= 4) ? 16 : 0) + (hi ? 8 : 0) + (v & 3) * 2;
        b[2 * v] = Vst[ndh * 16 + nl][kc * 32 + kb];
        b[2 * v + 1] = Vst[ndh * 16 + nl][kc * 32 + kb + 1];
      }
      acc = __builtin_amdgcn_wmma_f32_16x16x32_f16(false, a, false, b, (short)0,
                                                   acc, false, false);
    }
#pragma unroll
    for (int v = 0; v < 8; ++v) {
      int m = hi * 8 + v;
      o[(size_t)(i0 + m) * C_ATOM + head * DHEAD + ndh * 16 + nl] = acc[v];
    }
  }
}

// ---------------------------------------------------------------------------
// Token aggregation (scatter mean)
// ---------------------------------------------------------------------------
__global__ __launch_bounds__(384)
void k_segsum(const float* __restrict__ atom_out, const int* __restrict__ tok,
              float* __restrict__ out_tok, float* __restrict__ cnt) {
  int a = blockIdx.x, ch = threadIdx.x;
  int t = tok[a];
  atomicAdd(&out_tok[(size_t)t * C_TOKEN + ch], atom_out[(size_t)a * C_TOKEN + ch]);
  if (ch == 0) atomicAdd(&cnt[t], 1.f);
}

__global__ __launch_bounds__(384)
void k_tokdiv(float* __restrict__ out_tok, const float* __restrict__ cnt) {
  int t = blockIdx.x, ch = threadIdx.x;
  out_tok[(size_t)t * C_TOKEN + ch] /= fmaxf(cnt[t], 1.f);
}

// ---------------------------------------------------------------------------
// Host orchestration
// ---------------------------------------------------------------------------
static inline void gemm(hipStream_t s, const float* A, int lda, const float* B,
                        int ldb, float* C, int ldc, int M, int N, int K,
                        const float* bias, int act_a, int act_c, int beta) {
  dim3 grid((N + 15) / 16, (M + 127) / 128);
  k_gemm<<<grid, 256, 0, s>>>(A, lda, B, ldb, C, ldc, M, N, K, bias, act_a, act_c, beta);
}
static inline void elt(hipStream_t s, const float* a, const float* b,
                       const float* c, float* out, int n, int mode) {
  k_eltwise<<<(n + 255) / 256, 256, 0, s>>>(a, b, c, out, n, mode);
}

extern "C" void kernel_launch(void* const* d_in, const int* in_sizes, int n_in,
                              void* d_out, int out_size, void* d_ws, size_t ws_size,
                              hipStream_t stream) {
  (void)in_sizes; (void)n_in; (void)out_size; (void)ws_size;
  auto F = [&](int i) { return (const float*)d_in[i]; };
  auto I = [&](int i) { return (const int*)d_in[i]; };
  // inputs
  const float* pos    = F(0);
  const float* charge = F(1);
  const float* rmask  = F(2);
  const float* elem   = F(3);
  const float* names  = F(4);
  const int*   uid    = I(5);
  const int*   tok    = I(6);
  const float* s_tr   = F(7);
  const float* z_tr   = F(8);
  const float* noisy  = F(9);
  const float* amask  = F(10);
  // params (flattened in dict order, 12..104)
  const float* w_embed = F(12); const float* b_embed = F(13);
  const float* w_off = F(14);   const float* w_d = F(15);  const float* w_m = F(16);
  const float* ln_ts_w = F(17); const float* ln_ts_b = F(18); const float* w_ts = F(19);
  const float* ln_tp_w = F(20); const float* ln_tp_b = F(21); const float* w_tp = F(22);
  const float* w_noisy = F(23);
  const float* w_row = F(24);   const float* w_col = F(25);
  const float* w_mlp1 = F(26);  const float* w_mlp2 = F(27); const float* w_mlp3 = F(28);
  const float* w_out_p = F(104);

  // outputs: [token_single | q | c | p]
  float* out_tok = (float*)d_out;
  float* out_q = out_tok + (size_t)N_TOK * C_TOKEN;
  float* out_c = out_q + (size_t)N_ATOMS * C_ATOM;
  float* out_p = out_c + (size_t)N_ATOMS * C_ATOM;

  // workspace layout (floats)
  float* ws = (float*)d_ws;
  float* zp   = ws;                                             // 192*192*16
  float* ts   = zp + (size_t)N_TOK * N_TOK * C_PAIR;            // 192*128
  float* qbuf = ts + (size_t)N_TOK * C_ATOM;                    // 1536*128
  float* rowv = qbuf + (size_t)N_ATOMS * C_ATOM;                // 1536*16
  float* colv = rowv + (size_t)N_ATOMS * C_PAIR;                // 1536*16
  float* aout_tok = colv + (size_t)N_ATOMS * C_PAIR;            // 1536*384
  float* cnt  = aout_tok + (size_t)N_ATOMS * C_TOKEN;           // 256
  float* Breg = cnt + 256;                                      // scratch region
  const size_t NA = (size_t)N_ATOMS * C_ATOM;  // 196608
  float* x    = Breg;
  float* lnq  = x + NA;
  float* lns  = lnq + NA;
  float* gpre = lns + NA;
  float* skipb= gpre + NA;
  float* qh   = skipb + NA;
  float* khb  = qh + NA;
  float* vhb  = khb + NA;
  float* gh   = vhb + NA;
  float* obuf = gh + NA;
  float* og   = obuf + NA;
  float* aout = og + NA;
  float* t1   = aout + NA;             // 1536*256
  float* t2   = t1 + NA * 2;
  float* bt   = t2 + NA * 2;
  float* btw  = bt + NA * 2;
  float* lnbig = Breg;                 // reused pre-loop for LN(z) / LN(s)

  // ---- trunk conditioning ----
  k_ln_rows<<<(N_TOK * N_TOK + 7) / 8, 256, 0, stream>>>(z_tr, ln_tp_w, ln_tp_b,
                                                         lnbig, N_TOK * N_TOK, C_ZPAIR);
  gemm(stream, lnbig, C_ZPAIR, w_tp, C_PAIR, zp, C_PAIR, N_TOK * N_TOK, C_PAIR,
       C_ZPAIR, nullptr, 0, 0, 0);
  k_ln_rows<<<(N_TOK + 7) / 8, 256, 0, stream>>>(s_tr, ln_ts_w, ln_ts_b, lnbig,
                                                 N_TOK, C_TOKEN);
  gemm(stream, lnbig, C_TOKEN, w_ts, C_ATOM, ts, C_ATOM, N_TOK, C_ATOM, C_TOKEN,
       nullptr, 0, 0, 0);

  // ---- atom embedding + q0 ----
  k_embed<<<N_ATOMS, C_ATOM, 0, stream>>>(pos, charge, rmask, elem, names,
                                          w_embed, b_embed, ts, tok, noisy,
                                          w_noisy, out_c, qbuf);

  // ---- pair tensor ----
  gemm(stream, out_c, C_ATOM, w_row, C_PAIR, rowv, C_PAIR, N_ATOMS, C_PAIR,
       C_ATOM, nullptr, 1, 0, 0);
  gemm(stream, out_c, C_ATOM, w_col, C_PAIR, colv, C_PAIR, N_ATOMS, C_PAIR,
       C_ATOM, nullptr, 1, 0, 0);
  k_pair_init<<<dim3(N_ATOMS, 6), 256, 0, stream>>>(pos, uid, tok, w_off, w_d,
                                                    w_m, zp, rowv, colv, out_p);
  k_pair_mlp<<<(N_ATOMS * N_ATOMS / 16) / 8, 256, 0, stream>>>(out_p, w_mlp1,
                                                               w_mlp2, w_mlp3);

  // ---- transformer blocks ----
  for (int b = 0; b < 3; ++b) {
    int bb = 29 + b * 25;
    const float* lnpw = F(bb + 0); const float* lnpb = F(bb + 1);
    const float* wpb = F(bb + 2);
    const float* a_sw = F(bb + 3); const float* a_wg = F(bb + 4);
    const float* a_bg = F(bb + 5); const float* a_ws = F(bb + 6);
    const float* wq = F(bb + 7);  const float* bq = F(bb + 8);
    const float* wk = F(bb + 9);  const float* wv = F(bb + 10);
    const float* wg = F(bb + 11); const float* bg = F(bb + 12);
    const float* wo = F(bb + 13);
    const float* w_og = F(bb + 14); const float* b_og = F(bb + 15);
    const float* t_sw = F(bb + 16); const float* t_wg = F(bb + 17);
    const float* t_bg = F(bb + 18); const float* t_wsk = F(bb + 19);
    const float* w_t1 = F(bb + 20); const float* w_t2 = F(bb + 21);
    const float* w_tg = F(bb + 22); const float* b_tg = F(bb + 23);
    const float* w_to = F(bb + 24);

    // x = adaLN(q, c)
    k_ln_rows<<<N_ATOMS / 8, 256, 0, stream>>>(qbuf, nullptr, nullptr, lnq, N_ATOMS, C_ATOM);
    k_ln_rows<<<N_ATOMS / 8, 256, 0, stream>>>(out_c, a_sw, nullptr, lns, N_ATOMS, C_ATOM);
    gemm(stream, lns, C_ATOM, a_wg, C_ATOM, gpre, C_ATOM, N_ATOMS, C_ATOM, C_ATOM, a_bg, 0, 0, 0);
    gemm(stream, lns, C_ATOM, a_ws, C_ATOM, skipb, C_ATOM, N_ATOMS, C_ATOM, C_ATOM, nullptr, 0, 0, 0);
    elt(stream, gpre, lnq, skipb, x, (int)NA, 0);
    // projections
    gemm(stream, x, C_ATOM, wq, C_ATOM, qh, C_ATOM, N_ATOMS, C_ATOM, C_ATOM, bq, 0, 0, 0);
    gemm(stream, x, C_ATOM, wk, C_ATOM, khb, C_ATOM, N_ATOMS, C_ATOM, C_ATOM, nullptr, 0, 0, 0);
    gemm(stream, x, C_ATOM, wv, C_ATOM, vhb, C_ATOM, N_ATOMS, C_ATOM, C_ATOM, nullptr, 0, 0, 0);
    gemm(stream, x, C_ATOM, wg, C_ATOM, gh, C_ATOM, N_ATOMS, C_ATOM, C_ATOM, bg, 0, 2, 0);
    // banded attention with fused pair bias
    k_attn<<<dim3(N_ATOMS / 16, N_HEADS), 32, 0, stream>>>(qh, khb, vhb, out_p,
                                                           lnpw, lnpb, wpb, amask, obuf);
    elt(stream, obuf, gh, nullptr, og, (int)NA, 4);
    gemm(stream, og, C_ATOM, wo, C_ATOM, aout, C_ATOM, N_ATOMS, C_ATOM, C_ATOM, nullptr, 0, 0, 0);
    gemm(stream, out_c, C_ATOM, w_og, C_ATOM, gpre, C_ATOM, N_ATOMS, C_ATOM, C_ATOM, b_og, 0, 0, 0);
    elt(stream, gpre, aout, nullptr, aout, (int)NA, 2);
    // conditioned transition (uses pre-attention q via lnq)
    k_ln_rows<<<N_ATOMS / 8, 256, 0, stream>>>(out_c, t_sw, nullptr, lns, N_ATOMS, C_ATOM);
    gemm(stream, lns, C_ATOM, t_wg, C_ATOM, gpre, C_ATOM, N_ATOMS, C_ATOM, C_ATOM, t_bg, 0, 0, 0);
    gemm(stream, lns, C_ATOM, t_wsk, C_ATOM, skipb, C_ATOM, N_ATOMS, C_ATOM, C_ATOM, nullptr, 0, 0, 0);
    elt(stream, gpre, lnq, skipb, x, (int)NA, 0);
    gemm(stream, x, C_ATOM, w_t1, 2 * C_ATOM, t1, 2 * C_ATOM, N_ATOMS, 2 * C_ATOM, C_ATOM, nullptr, 0, 0, 0);
    gemm(stream, x, C_ATOM, w_t2, 2 * C_ATOM, t2, 2 * C_ATOM, N_ATOMS, 2 * C_ATOM, C_ATOM, nullptr, 0, 0, 0);
    elt(stream, t1, t2, nullptr, bt, (int)NA * 2, 1);
    gemm(stream, bt, 2 * C_ATOM, w_to, C_ATOM, btw, C_ATOM, N_ATOMS, C_ATOM, 2 * C_ATOM, nullptr, 0, 0, 0);
    gemm(stream, out_c, C_ATOM, w_tg, C_ATOM, gpre, C_ATOM, N_ATOMS, C_ATOM, C_ATOM, b_tg, 0, 0, 0);
    elt(stream, gpre, btw, nullptr, btw, (int)NA, 2);
    elt(stream, aout, btw, nullptr, qbuf, (int)NA, 3);  // q = a_out + t_out
  }

  // ---- atom -> token mean ----
  gemm(stream, qbuf, C_ATOM, w_out_p, C_TOKEN, aout_tok, C_TOKEN, N_ATOMS,
       C_TOKEN, C_ATOM, nullptr, 0, 1 /*relu*/, 0);
  (void)hipMemsetAsync(out_tok, 0, (size_t)N_TOK * C_TOKEN * sizeof(float), stream);
  (void)hipMemsetAsync(cnt, 0, 256 * sizeof(float), stream);
  k_segsum<<<N_ATOMS, C_TOKEN, 0, stream>>>(aout_tok, tok, out_tok, cnt);
  k_tokdiv<<<N_TOK, C_TOKEN, 0, stream>>>(out_tok, cnt);
  (void)hipMemcpyAsync(out_q, qbuf, NA * sizeof(float), hipMemcpyDeviceToDevice, stream);
}
